// MultiHeadLatentAttention_75926431858945
// MI455X (gfx1250) — compile-verified, matrix operand order checked by reading, not run on hardware
//
#include <hip/hip_runtime.h>

typedef __attribute__((ext_vector_type(16))) __bf16 v16bf;
typedef __attribute__((ext_vector_type(8)))  __bf16 v8bf;
typedef __attribute__((ext_vector_type(4)))  __bf16 v4bf;
typedef __attribute__((ext_vector_type(8)))  float  v8f;

#define NQ   2048
#define NKV_ 2048
#define EMB  4096
#define NH   16
#define HD   256
#define LQ_  512
#define LKV_ 256

// CDNA5 async global->LDS copy path (ASYNCcnt-tracked), guarded so either
// toolchain compiles; fallback is a plain vector copy through VGPRs.
#if __has_builtin(__builtin_amdgcn_global_load_async_to_lds_b128) && \
    __has_builtin(__builtin_amdgcn_s_wait_asynccnt)
#define HAVE_ASYNC_LDS 1
typedef int v4i_async __attribute__((vector_size(16)));                 // int4
typedef __attribute__((address_space(1))) v4i_async* gas_ptr;           // global src
typedef __attribute__((address_space(3))) v4i_async* las_ptr;           // LDS dst
#else
#define HAVE_ASYNC_LDS 0
#endif

__device__ __forceinline__ void async_copy_b128(const __bf16* g, __bf16* l) {
#if HAVE_ASYNC_LDS
  __builtin_amdgcn_global_load_async_to_lds_b128((gas_ptr)g, (las_ptr)l, 0, 0);
#else
  *(v8bf*)l = *(const v8bf*)g;
#endif
}
__device__ __forceinline__ void async_wait0() {
#if HAVE_ASYNC_LDS
  __builtin_amdgcn_s_wait_asynccnt(0);
#endif
}

__device__ __forceinline__ __bf16 f2bf(float f) {
  unsigned u = __builtin_bit_cast(unsigned, f);
  unsigned r = (u + 0x7FFFu + ((u >> 16) & 1u)) >> 16;  // round-to-nearest-even
  unsigned short s = (unsigned short)r;
  return __builtin_bit_cast(__bf16, s);
}

__device__ __forceinline__ v8f zero8() {
  v8f z;
  #pragma unroll
  for (int i = 0; i < 8; ++i) z[i] = 0.0f;
  return z;
}

// ---------------------------------------------------------------------------
// Tiled bf16 WMMA GEMM: C[M,N](bf16) = A[M,K] * B[K,N](f32), row-major.
// Block tile 128x128, K-step 32. 8 waves: 4 row-groups x 2 col-groups,
// each wave computes 32x64 via 2x4 tiles of v_wmma_f32_16x16x32_bf16.
// f32 A/B are converted to bf16 in the load stage (b128 global loads);
// bf16 A tiles go straight to LDS via async b128 copies.
// ---------------------------------------------------------------------------
template <typename TA>
__global__ __launch_bounds__(256)
void gemm_bf16(const TA* __restrict__ A, const float* __restrict__ B,
               __bf16* __restrict__ C, int M, int N, int K) {
  __shared__ __attribute__((aligned(32))) __bf16 As[128 * 32];   // [m][k]
  __shared__ __attribute__((aligned(32))) __bf16 Bt[128 * 32];   // transposed [n][k]
  const int tid  = threadIdx.x;
  const int lane = tid & 31;
  const int wave = tid >> 5;
  const int wr   = wave & 3;   // row group (4 x 32 rows)
  const int wc   = wave >> 2;  // col group (2 x 64 cols)
  const int m0   = blockIdx.y * 128;
  const int n0   = blockIdx.x * 128;
  (void)M;

  v8f acc[2][4];
  #pragma unroll
  for (int i = 0; i < 2; ++i)
    #pragma unroll
    for (int j = 0; j < 4; ++j) acc[i][j] = zero8();

  for (int k0 = 0; k0 < K; k0 += 32) {
    // ---- A tile 128x32 -> LDS ----
    if constexpr (sizeof(TA) == 2) {
      // bf16 A: pure copy, use async global->LDS b128 (2 chunks/thread)
      #pragma unroll
      for (int j = 0; j < 2; ++j) {
        int e = tid + j * 256;          // 512 chunks of 8 elems
        int r = e >> 2, c = (e & 3) * 8;
        async_copy_b128((const __bf16*)&A[(size_t)(m0 + r) * K + k0 + c],
                        &As[r * 32 + c]);
      }
    } else {
      // f32 A: b128 loads, convert, 8-byte packed LDS stores (4 chunks/thread)
      #pragma unroll
      for (int j = 0; j < 4; ++j) {
        int e = tid + j * 256;          // 1024 chunks of 4 elems
        int r = e >> 3, c = (e & 7) * 4;
        float4 f = *(const float4*)&A[(size_t)(m0 + r) * K + k0 + c];
        v4bf p;
        p[0] = f2bf(f.x); p[1] = f2bf(f.y); p[2] = f2bf(f.z); p[3] = f2bf(f.w);
        *(v4bf*)&As[r * 32 + c] = p;
      }
    }
    // ---- B tile 32x128 -> LDS transposed [n][k] (b128 loads, scalar stores)
    #pragma unroll
    for (int j = 0; j < 4; ++j) {
      int e = tid + j * 256;            // 1024 chunks of 4 elems
      int r = e >> 5, c = (e & 31) * 4;
      float4 f = *(const float4*)&B[(size_t)(k0 + r) * N + n0 + c];
      Bt[(c + 0) * 32 + r] = f2bf(f.x);
      Bt[(c + 1) * 32 + r] = f2bf(f.y);
      Bt[(c + 2) * 32 + r] = f2bf(f.z);
      Bt[(c + 3) * 32 + r] = f2bf(f.w);
    }
    async_wait0();
    __syncthreads();

    v16bf af[2], bfr[4];
    #pragma unroll
    for (int mt = 0; mt < 2; ++mt)
      af[mt] = *(const v16bf*)&As[(wr * 32 + mt * 16 + (lane & 15)) * 32 + (lane >> 4) * 16];
    #pragma unroll
    for (int nt = 0; nt < 4; ++nt)
      bfr[nt] = *(const v16bf*)&Bt[(wc * 64 + nt * 16 + (lane & 15)) * 32 + (lane >> 4) * 16];

    #pragma unroll
    for (int mt = 0; mt < 2; ++mt)
      #pragma unroll
      for (int nt = 0; nt < 4; ++nt)
        acc[mt][nt] = __builtin_amdgcn_wmma_f32_16x16x32_bf16(
            false, af[mt], false, bfr[nt], (short)0, acc[mt][nt], false, false);
    __syncthreads();
  }

  // Epilogue per ISA 16x16 f32 C layout: VGPR i, lanes0-15 -> M=i, lanes16-31 -> M=i+8
  #pragma unroll
  for (int mt = 0; mt < 2; ++mt)
    #pragma unroll
    for (int nt = 0; nt < 4; ++nt)
      #pragma unroll
      for (int i = 0; i < 8; ++i) {
        int row = m0 + wr * 32 + mt * 16 + i + ((lane < 16) ? 0 : 8);
        int col = n0 + wc * 64 + nt * 16 + (lane & 15);
        C[(size_t)row * N + col] = f2bf(acc[mt][nt][i]);
      }
}

// ---------------------------------------------------------------------------
// Flash-attention over the latent dim.
//   Qh  = CqWqk [B, N, H, LKV]   (bf16)  -- absorbed queries
//   Ckv = [B, NKV, LKV]          (bf16)  -- shared K latent
//   Vup = [B, NKV, H, HD]        (bf16)
//   Out = [B, N, H*HD]           (f32)
// Grid (N/64, H, B); 8 waves: 4 row-groups (16 q rows) x 2 col-groups (128 hd)
// ---------------------------------------------------------------------------
__global__ __launch_bounds__(256)
void mla_flash(const __bf16* __restrict__ Qh, const __bf16* __restrict__ Ckv,
               const __bf16* __restrict__ Vup, float* __restrict__ Out) {
  __shared__ __attribute__((aligned(32))) __bf16 Kt[32 * LKV_];   // [kv][lkv] 16KB
  __shared__ __attribute__((aligned(32))) __bf16 Vt[HD * 32];     // [hd][kv]  16KB
  __shared__ __attribute__((aligned(32))) __bf16 Ps[8 * 16 * 32]; // P staging  8KB

  const int tid  = threadIdx.x;
  const int lane = tid & 31;
  const int wave = tid >> 5;
  const int wr   = wave & 3;
  const int wc   = wave >> 2;
  const int bb   = blockIdx.z;
  const int h    = blockIdx.y;
  const int row0 = blockIdx.x * 64 + wr * 16;
  const int c0   = wc * 128;

  // Q fragments for this wave's 16 rows: 16 x 256 over 8 k-steps of 32
  v16bf qf[8];
  {
    const size_t qbase = ((size_t)(bb * NQ + row0 + (lane & 15)) * NH + h) * LKV_;
    #pragma unroll
    for (int ks = 0; ks < 8; ++ks)
      qf[ks] = *(const v16bf*)&Qh[qbase + ks * 32 + (lane >> 4) * 16];
  }

  v8f o[8];
  #pragma unroll
  for (int t = 0; t < 8; ++t) o[t] = zero8();
  float mrow[8], lrow[8];
  #pragma unroll
  for (int i = 0; i < 8; ++i) { mrow[i] = -1e30f; lrow[i] = 0.0f; }

  __bf16* myP = &Ps[wave * 512];

  for (int kv0 = 0; kv0 < NKV_; kv0 += 32) {
    // K tile [32][256] bf16: straight copy -> async global->LDS b128
    #pragma unroll
    for (int j = 0; j < 4; ++j) {
      int e = tid + j * 256;            // 1024 chunks of 8 elems (16B)
      int kvr = e >> 5, c8 = (e & 31) * 8;
      async_copy_b128(&Ckv[(size_t)(bb * NKV_ + kv0 + kvr) * LKV_ + c8],
                      &Kt[kvr * LKV_ + c8]);
    }
    // V tile transposed [256][32]: 32B global loads + scalar LDS stores
    #pragma unroll
    for (int j = 0; j < 2; ++j) {
      int e = tid + j * 256;            // 512 chunks of 16 elems
      int kvr = e >> 4, cc = e & 15;
      v16bf v = *(const v16bf*)&Vup[((size_t)(bb * NKV_ + kv0 + kvr) * NH + h) * HD + cc * 16];
      #pragma unroll
      for (int jj = 0; jj < 16; ++jj)
        Vt[(cc * 16 + jj) * 32 + kvr] = v[jj];
    }
    if (kv0 + 32 < NKV_) {
      __builtin_prefetch(&Ckv[(size_t)(bb * NKV_ + kv0 + 32) * LKV_], 0, 1);
      __builtin_prefetch(&Vup[((size_t)(bb * NKV_ + kv0 + 32) * NH + h) * HD], 0, 1);
    }
    async_wait0();
    __syncthreads();

    // S = Q * K^T for 16 rows x 32 kv cols (two 16x16 accum tiles)
    v8f s0 = zero8(), s1 = zero8();
    #pragma unroll
    for (int ks = 0; ks < 8; ++ks) {
      v16bf b0 = *(const v16bf*)&Kt[(0  + (lane & 15)) * LKV_ + ks * 32 + (lane >> 4) * 16];
      v16bf b1 = *(const v16bf*)&Kt[(16 + (lane & 15)) * LKV_ + ks * 32 + (lane >> 4) * 16];
      s0 = __builtin_amdgcn_wmma_f32_16x16x32_bf16(false, qf[ks], false, b0, (short)0, s0, false, false);
      s1 = __builtin_amdgcn_wmma_f32_16x16x32_bf16(false, qf[ks], false, b1, (short)0, s1, false, false);
    }

    // Online softmax; rows live in 16-lane groups per the C-matrix layout
    const float scale = 0.0625f;  // 1/sqrt(LKV=256)
    #pragma unroll
    for (int i = 0; i < 8; ++i) {
      float a0 = s0[i] * scale;
      float a1 = s1[i] * scale;
      float v = fmaxf(a0, a1);
      #pragma unroll
      for (int off = 8; off >= 1; off >>= 1)
        v = fmaxf(v, __shfl_xor(v, off, 32));
      float mnew  = fmaxf(mrow[i], v);
      float alpha = __expf(mrow[i] - mnew);
      float p0 = __expf(a0 - mnew);
      float p1 = __expf(a1 - mnew);
      float ps = p0 + p1;
      #pragma unroll
      for (int off = 8; off >= 1; off >>= 1)
        ps += __shfl_xor(ps, off, 32);
      lrow[i] = lrow[i] * alpha + ps;
      mrow[i] = mnew;
      #pragma unroll
      for (int t = 0; t < 8; ++t) o[t][i] *= alpha;
      int prow = i + ((lane < 16) ? 0 : 8);
      myP[prow * 32 + (lane & 15)]      = f2bf(p0);
      myP[prow * 32 + 16 + (lane & 15)] = f2bf(p1);
    }
    asm volatile("s_wait_dscnt 0" ::: "memory");  // wave-private LDS RAW

    // O += P * V  (contraction over 32 kv -> one WMMA per 16-col output tile)
    v16bf pa = *(const v16bf*)&myP[(lane & 15) * 32 + (lane >> 4) * 16];
    #pragma unroll
    for (int t = 0; t < 8; ++t) {
      v16bf bv = *(const v16bf*)&Vt[(c0 + t * 16 + (lane & 15)) * 32 + (lane >> 4) * 16];
      o[t] = __builtin_amdgcn_wmma_f32_16x16x32_bf16(false, pa, false, bv, (short)0, o[t], false, false);
    }
    __syncthreads();
  }

  // Normalize and write [B,N,H*HD] f32
  #pragma unroll
  for (int i = 0; i < 8; ++i) {
    float inv = (lrow[i] > 0.0f) ? (1.0f / lrow[i]) : 0.0f;
    int row = row0 + i + ((lane < 16) ? 0 : 8);
    #pragma unroll
    for (int t = 0; t < 8; ++t) {
      int col = h * HD + c0 + t * 16 + (lane & 15);
      Out[(size_t)(bb * NQ + row) * EMB + col] = o[t][i] * inv;
    }
  }
}

// ---------------------------------------------------------------------------
extern "C" void kernel_launch(void* const* d_in, const int* in_sizes, int n_in,
                              void* d_out, int out_size, void* d_ws, size_t ws_size,
                              hipStream_t stream) {
  (void)in_sizes; (void)n_in; (void)out_size; (void)ws_size;
  const float* Q     = (const float*)d_in[0];   // [B,N,E]
  const float* K     = (const float*)d_in[1];   // [B,NKV,E]
  const float* Wq_d  = (const float*)d_in[2];   // [E,LQ]
  const float* W_qk  = (const float*)d_in[3];   // [LQ,H*LKV]
  const float* Wkv_d = (const float*)d_in[4];   // [E,LKV]
  const float* Wv_u  = (const float*)d_in[5];   // [LKV,H*HD]
  float* out = (float*)d_out;

  char* ws = (char*)d_ws;
  __bf16* C_q    = (__bf16*)(ws);                                  //  4 MiB [4096,512]
  __bf16* C_kv   = (__bf16*)(ws + ((size_t)4  << 20));             //  2 MiB [4096,256]
  __bf16* CqWqk  = (__bf16*)(ws + ((size_t)8  << 20));             // 32 MiB [4096,4096]
  __bf16* V_up   = (__bf16*)(ws + ((size_t)40 << 20));             // 32 MiB [4096,4096]

  dim3 blk(256);
  // C_q   = Q @ Wq_d        [4096,4096] x [4096,512]
  gemm_bf16<float><<<dim3(LQ_ / 128, (2 * NQ) / 128), blk, 0, stream>>>(
      Q, Wq_d, C_q, 2 * NQ, LQ_, EMB);
  // C_kv  = K @ Wkv_d       [4096,4096] x [4096,256]
  gemm_bf16<float><<<dim3(LKV_ / 128, (2 * NKV_) / 128), blk, 0, stream>>>(
      K, Wkv_d, C_kv, 2 * NKV_, LKV_, EMB);
  // CqWqk = C_q @ W_qk      [4096,512] x [512,4096]
  gemm_bf16<__bf16><<<dim3(EMB / 128, (2 * NQ) / 128), blk, 0, stream>>>(
      C_q, W_qk, CqWqk, 2 * NQ, EMB, LQ_);
  // V_up  = C_kv @ Wv_u     [4096,256] x [256,4096]
  gemm_bf16<__bf16><<<dim3(EMB / 128, (2 * NKV_) / 128), blk, 0, stream>>>(
      C_kv, Wv_u, V_up, 2 * NKV_, EMB, LKV_);
  // Flash attention over latent dim
  mla_flash<<<dim3(NQ / 64, NH, 2), blk, 0, stream>>>(CqWqk, C_kv, V_up, out);
}
